// KernelNCA_13907104104662
// MI455X (gfx1250) — matrix operand order repaired and checked by reading, not compiled, not run on hardware
//
#include <hip/hip_runtime.h>
#include <hip/hip_bf16.h>
#include <math.h>

typedef __attribute__((ext_vector_type(16))) _Float16 v16h;
typedef __attribute__((ext_vector_type(8)))  float    v8f;
typedef __attribute__((ext_vector_type(4)))  unsigned int v4u;
typedef __attribute__((ext_vector_type(8)))  int      v8i;
typedef __attribute__((ext_vector_type(4)))  int      v4i;

#define NQ     512
#define NC     8000
#define DNUM   6
#define DCAT   20
#define NBINS  50
#define DENC   (DNUM * NBINS)   /* 300 */
#define DTOT   (DENC + DCAT)    /* 320 */
#define D4     (DTOT / 4)       /* 80  */
#define DOUT   10
#define CHUNK  1000
#define NCHUNK (NC / CHUNK)     /* 8   */
#define KTILES 32               /* ceil(1000/32), last tile has 8 valid */
#define NSUB   4                /* candidate sub-splits per chunk        */
#define TPS    (KTILES / NSUB)  /* 8 k-tiles per block                   */
#define TILEF  (32 * DTOT)      /* floats per LDS tile buffer            */

#ifndef __has_builtin
#define __has_builtin(x) 0
#endif

#if __has_builtin(__builtin_amdgcn_tensor_load_to_lds) && \
    __has_builtin(__builtin_amdgcn_s_wait_tensorcnt)
#define USE_TDM 1
#else
#define USE_TDM 0
#endif

// ---------------------------------------------------------------------------
// Encode: out[row][j] = j<300 ? ceil((x_num[row][j/50] - u*delta)/delta)
//                             : x_cat[row][j-300]
// ---------------------------------------------------------------------------
__global__ __launch_bounds__(256) void nca_encode(
    const float* __restrict__ xnum, const float* __restrict__ xcat,
    const float* __restrict__ delta, const float* __restrict__ u,
    float* __restrict__ enc, int nrows)
{
  int idx = blockIdx.x * blockDim.x + threadIdx.x;
  int total = nrows * DTOT;
  if (idx >= total) return;
  int row = idx / DTOT, j = idx - row * DTOT;
  float v;
  if (j < DENC) {
    int i = j / NBINS, b = j - i * NBINS;
    float dl = delta[i * NBINS + b];
    float uu = u[i * NBINS + b];
    v = ceilf((xnum[row * DNUM + i] - uu * dl) / dl);
  } else {
    v = xcat[row * DCAT + (j - DENC)];
  }
  enc[idx] = v;
}

// ---------------------------------------------------------------------------
// Main: block = (32-query tile) x (256-candidate (8 k-tile) sub-range of one chunk).
// Double-buffered TDM staging of 32x320-f32 candidate tiles into LDS,
// L1 dists -> exp -> v_wmma_f32_16x16x32_f16 with one-hot labels.
// Emits additive logits partials + raw (m, s) online-softmax partials.
// ---------------------------------------------------------------------------
__global__ __launch_bounds__(64) void nca_main(
    const float* __restrict__ xenc,   // [NQ][DTOT]
    const float* __restrict__ cenc,   // [NC][DTOT]
    const int*   __restrict__ cy,     // [NC]
    float* __restrict__ plogits,      // [NCHUNK*NSUB][NQ][16]
    float* __restrict__ pm,           // [NCHUNK*NSUB][NQ]  (running max)
    float* __restrict__ ps)           // [NCHUNK*NSUB][NQ]  (scaled sum)
{
  extern __shared__ float ct[];       // 2 tiles x 32 x 320 f32 = 80 KB

  const int chunk  = blockIdx.y >> 2;       // blockIdx.y = chunk*NSUB + sub
  const int sub    = blockIdx.y & (NSUB - 1);
  const int chunk0 = chunk * CHUNK;
  const int qbase  = blockIdx.x * 32;
  const int tid    = threadIdx.x;
  const int lane   = tid & 31;
  const int wave   = tid >> 5;
  const int qrow   = qbase + wave * 16 + (lane & 15);  // A-matrix row M
  const float* __restrict__ xrow = xenc + (size_t)qrow * DTOT;
  const int hiK  = (lane < 16) ? 0 : 8;   // A layout: lanes>=16 hold K+8
  const int ncol = lane & 15;             // B/C column N
  const int kb   = (lane < 16) ? 0 : 16;  // unused; B uses hiK, same k order as A

  v8f acc = {};                 // 16x16 f32 C/D accumulator (logits tile)
  float m = -1e30f, ssum = 0.f; // online softmax partials for this lane

#if USE_TDM
  // Tensor Data Mover: 2D tile (32 rows x 320 f32) -> LDS buffer (t&1).
  // Rows beyond tensor_dim1 = NC - cbase are zero-filled by the TDM.
  auto stage = [&](int t) {
    if (wave != 0) return;
    int cbase = chunk0 + (sub * TPS + t) * 32;
    unsigned long long ga =
        (unsigned long long)(const char*)cenc + (unsigned long long)cbase * (DTOT * 4);
    unsigned ldsa  = (unsigned)(unsigned long long)(void*)(ct + (size_t)(t & 1) * TILEF);
    unsigned tdim1 = (unsigned)(NC - cbase);
    // D# group 0: count=1 | lds_addr | global_addr[31:0] | ga[56:32], type=2
    v4u g0 = { 1u, ldsa, (unsigned)(ga & 0xFFFFFFFFu),
               (unsigned)((ga >> 32) & 0x01FFFFFFu) | 0x80000000u };
    // D# group 1: data_size=4B; tdim0=320; tdim1; tile0=320; tile1=32;
    //             dim0_stride=320; dims 2+ unused.
    v8i g1 = { (int)0x20000u,                    // wg_mask=0, data_size=4B
               (int)(320u << 16),                // tensor_dim0[15:0] @ bit48
               (int)((tdim1 & 0xFFFFu) << 16),   // tdim0 hi=0 | tensor_dim1[15:0]
               (int)(320u << 16),                // tdim1 hi=0 | tile_dim0=320
               32,                               // tile_dim1=32 | tile_dim2=0
               320,                              // tensor_dim0_stride lo
               0, 0 };                           // stride hi / dim1_stride
    v4i gz = { 0, 0, 0, 0 };
#if __clang_major__ >= 23
    v8i gz8 = { 0, 0, 0, 0, 0, 0, 0, 0 };
    __builtin_amdgcn_tensor_load_to_lds(g0, g1, gz, gz, gz8, 0);
#else
    __builtin_amdgcn_tensor_load_to_lds(g0, g1, gz, gz, 0);
#endif
  };
#define WAIT_STAGE() do { if (wave == 0) __builtin_amdgcn_s_wait_tensorcnt(0); } while (0)
#else
  // Fallback: contiguous per-thread synchronous copy into buffer (t&1).
  auto stage = [&](int t) {
    int cbase = chunk0 + (sub * TPS + t) * 32;
    float* dbuf = ct + (size_t)(t & 1) * TILEF;
    int c  = tid >> 1;
    int d0 = (tid & 1) * (D4 / 2);
    int g  = cbase + c; if (g > NC - 1) g = NC - 1;
    const float4* src = (const float4*)cenc + (size_t)g * D4 + d0;
    float4*       dst = (float4*)dbuf + c * D4 + d0;
    #pragma unroll 8
    for (int k = 0; k < D4 / 2; ++k) dst[k] = src[k];
  };
#define WAIT_STAGE() do { } while (0)
#endif

  stage(0);   // prologue: fill buffer 0

  for (int t = 0; t < TPS; ++t) {
    const int ktg   = sub * TPS + t;          // global k-tile in chunk (0..31)
    const int cbase = chunk0 + ktg * 32;
    const float* __restrict__ cb = ct + (size_t)(t & 1) * TILEF;

    WAIT_STAGE();        // tile t resident in LDS (TDM complete)
    __syncthreads();     // ...visible to both waves; prev compute done

    if (t + 1 < TPS) stage(t + 1);   // overlap next DMA with this compute

    // Prefetch tile t+2's source into L2 (global_prefetch_b8).
    if (t + 2 < TPS) {
      int gp = cbase + 64 + lane; if (gp > NC - 1) gp = NC - 1;
      __builtin_prefetch(cenc + (size_t)gp * DTOT, 0, 3);
    }

    // L1 distances: 16 (q, c) pairs per lane, matching the WMMA A layout.
    float dist[16];
    #pragma unroll
    for (int e = 0; e < 16; ++e) dist[e] = 0.f;
    for (int d4 = 0; d4 < D4; ++d4) {
      float4 xv = ((const float4*)xrow)[d4];
      #pragma unroll
      for (int e = 0; e < 16; ++e) {
        int cl = ((e < 8) ? e : e + 8) + hiK;           // K index for half e
        float4 cv = ((const float4*)cb)[cl * D4 + d4];  // broadcast in 16-lane group
        dist[e] += fabsf(xv.x - cv.x) + fabsf(xv.y - cv.y)
                 + fabsf(xv.z - cv.z) + fabsf(xv.w - cv.w);
      }
    }

    // A tile: exp(-dist) in f16; online stable softmax partials on the side.
    v16h av;
    #pragma unroll
    for (int e = 0; e < 16; ++e) {
      int cl = ((e < 8) ? e : e + 8) + hiK;
      bool valid = (ktg * 32 + cl) < CHUNK;
      float nd = -dist[e];
      float w = valid ? __expf(nd) : 0.f;
      av[e] = (_Float16)w;
      if (valid) {
        if (nd > m) { ssum = ssum * __expf(m - nd) + 1.f; m = nd; }
        else        { ssum += __expf(nd - m); }
      }
    }

    // B tile: one-hot labels, B[k][n] = (n == y[k]); zero for invalid k.
    v16h bv;
    #pragma unroll
    for (int e = 0; e < 16; ++e) {
      int c = ((e < 8) ? e : e + 8) + hiK;
      float b = 0.f;
      if (ktg * 32 + c < CHUNK) {
        int lab = cy[cbase + c];
        b = (ncol == lab) ? 1.f : 0.f;
      }
      bv[e] = (_Float16)b;
    }

    // logits_tile += exp(-dist_tile) @ onehot_tile
    acc = __builtin_amdgcn_wmma_f32_16x16x32_f16(
        /*neg_a=*/false, av, /*neg_b=*/false, bv,
        /*c_mod=*/(short)0, acc, /*reuse_a=*/false, /*reuse_b=*/false);
  }

  const int pidx = chunk * NSUB + sub;

  // Merge the two lane-halves of each query's (m, s) (lanes l and l^16).
  float m2 = __shfl_xor(m, 16);
  float s2 = __shfl_xor(ssum, 16);
  float M  = fmaxf(m, m2);
  float S  = ssum * __expf(m - M) + s2 * __expf(m2 - M);
  if (lane < 16) {
    pm[pidx * NQ + qrow] = M;
    ps[pidx * NQ + qrow] = S;
  }

  // C/D layout: vgpr r -> row M = r + (lane<16 ? 0 : 8), col N = lane%16.
  const int mofs = (lane < 16) ? 0 : 8;
  #pragma unroll
  for (int r = 0; r < 8; ++r) {
    int qq = qbase + wave * 16 + r + mofs;
    plogits[((size_t)pidx * NQ + qq) * 16 + ncol] = acc[r];
  }
}

// ---------------------------------------------------------------------------
// Finalize: fixed-order deterministic reduction. Per chunk, merge the NSUB
// (m, s) partials stably, then lse += M* + log(S*); logits are summed.
// ---------------------------------------------------------------------------
__global__ __launch_bounds__(256) void nca_finalize(
    const float* __restrict__ plogits, const float* __restrict__ pm,
    const float* __restrict__ ps, float* __restrict__ out)
{
  int idx = blockIdx.x * blockDim.x + threadIdx.x;
  if (idx >= NQ * DOUT) return;
  int q = idx / DOUT, n = idx - q * DOUT;

  float lse = 0.f;
  for (int ch = 0; ch < NCHUNK; ++ch) {
    float M = -1e30f;
    #pragma unroll
    for (int s = 0; s < NSUB; ++s)
      M = fmaxf(M, pm[(ch * NSUB + s) * NQ + q]);
    float S = 0.f;
    #pragma unroll
    for (int s = 0; s < NSUB; ++s)
      S += ps[(ch * NSUB + s) * NQ + q] * __expf(pm[(ch * NSUB + s) * NQ + q] - M);
    lse += M + __logf(S);
  }

  float lg = 0.f;
  #pragma unroll
  for (int p = 0; p < NCHUNK * NSUB; ++p)
    lg += plogits[((size_t)p * NQ + q) * 16 + n];

  out[idx] = __logf(lg + 1e-8f) - lse;
}

// ---------------------------------------------------------------------------
extern "C" void kernel_launch(void* const* d_in, const int* in_sizes, int n_in,
                              void* d_out, int out_size, void* d_ws, size_t ws_size,
                              hipStream_t stream) {
  const float* x_num = (const float*)d_in[0];   // [512][6]
  const float* x_cat = (const float*)d_in[1];   // [512][20]
  const float* c_num = (const float*)d_in[2];   // [8000][6]
  const float* c_cat = (const float*)d_in[3];   // [8000][20]
  const int*   c_y   = (const int*)  d_in[4];   // [8000]
  const float* delta = (const float*)d_in[5];   // [6][50]
  const float* u     = (const float*)d_in[6];   // [6][50]
  float* out = (float*)d_out;                   // [512][10]

  char* wsb = (char*)d_ws;
  float* xenc = (float*)(wsb);                                   //    655,360 B
  float* cenc = (float*)(wsb + 655360);                          // 10,240,000 B
  float* plg  = (float*)(wsb + 10895360);                        //  1,048,576 B
  float* pmv  = (float*)(wsb + 10895360 + 1048576);              //     65,536 B
  float* psv  = (float*)(wsb + 10895360 + 1048576 + 65536);      //     65,536 B

  nca_encode<<<(NQ * DTOT + 255) / 256, 256, 0, stream>>>(x_num, x_cat, delta, u, xenc, NQ);
  nca_encode<<<(NC * DTOT + 255) / 256, 256, 0, stream>>>(c_num, c_cat, delta, u, cenc, NC);

  dim3 grid(NQ / 32, NCHUNK * NSUB);   // 16 q-tiles x 32 chunk-subs = 512 blocks
  nca_main<<<grid, 64, 2 * TILEF * sizeof(float), stream>>>(xenc, cenc, c_y, plg, pmv, psv);

  nca_finalize<<<(NQ * DOUT + 255) / 256, 256, 0, stream>>>(plg, pmv, psv, out);
}
